// ContrastiveLoss_13408887899029
// MI455X (gfx1250) — matrix operand order repaired and checked by reading, not compiled, not run on hardware
//
#include <hip/hip_runtime.h>
#include <hip/hip_bf16.h>

// Problem constants (match reference: B=4096, D=256)
#define BN      4096          // batch
#define DDIM    256           // embedding dim
#define TWO_B   8192          // 2B rows
#define JSPLIT  8             // split of j (column) dimension across blocks.y
#define JTILES  ((TWO_B / JSPLIT) / 16)   // 64 j-tiles per block

// LDS B-tile: 16 rows x 256 halves, rows padded by 8 halves (16B) so that
// lane n's ds_load_b128 at row-stride 528B hits banks 4n..4n+3 (conflict-free).
#define BROW_H  264                       // halves per padded row (528 B)
#define BTILE_H (16 * BROW_H)             // 4224 halves = 8448 B per buffer

typedef __attribute__((ext_vector_type(16))) _Float16 v16h;
typedef __attribute__((ext_vector_type(8)))  _Float16 v8h;
typedef __attribute__((ext_vector_type(8)))  float    v8f;

// ---------------------------------------------------------------------------
// Kernel 1: L2-normalize rows of [emb_i; emb_j] -> f16 reps (row-major),
// pack (tag<<16)|doc metadata, zero-init denom, numer = 1. One wave32 per row.
// ---------------------------------------------------------------------------
__global__ void cl_normalize_kernel(const float* __restrict__ emb_i,
                                    const float* __restrict__ emb_j,
                                    const int* __restrict__ tags,
                                    const int* __restrict__ docs,
                                    _Float16* __restrict__ reps,
                                    unsigned* __restrict__ meta,
                                    float* __restrict__ denom,
                                    float* __restrict__ numer) {
    const int row  = (blockIdx.x * blockDim.x + threadIdx.x) >> 5;
    const int lane = threadIdx.x & 31;
    if (row >= TWO_B) return;

    const float* src = (row < BN) ? (emb_i + (size_t)row * DDIM)
                                  : (emb_j + (size_t)(row - BN) * DDIM);
    float v[DDIM / 32];
    float ss = 0.0f;
#pragma unroll
    for (int t = 0; t < DDIM / 32; ++t) {
        v[t] = src[lane + 32 * t];
        ss += v[t] * v[t];
    }
#pragma unroll
    for (int m = 16; m >= 1; m >>= 1) ss += __shfl_xor(ss, m, 32);
    const float inv = rsqrtf(ss);

    _Float16* dst = reps + (size_t)row * DDIM;
#pragma unroll
    for (int t = 0; t < DDIM / 32; ++t)
        dst[lane + 32 * t] = (_Float16)(v[t] * inv);

    if (lane == 0) {
        const int m = row & (BN - 1);     // tags2/docs2 = concat -> index % B
        meta[row]  = ((unsigned)tags[m] << 16) | (unsigned)docs[m];
        denom[row] = 0.0f;
        numer[row] = 1.0f;                // overwritten exactly once
    }
}

// ---------------------------------------------------------------------------
// Async copy of one contiguous 8KB B-tile (16 reps rows) into padded LDS.
// 256 threads x 2 x 16B chunks, via CDNA5 global_load_async_to_lds_b128
// (ASYNCcnt-tracked DMA; low 32 bits of a generic LDS pointer are the LDS
// byte offset the instruction's VDST expects).
// ---------------------------------------------------------------------------
__device__ __forceinline__ void issue_btile(const _Float16* __restrict__ gsrc,
                                            _Float16* lds_tile, int tid) {
#pragma unroll
    for (int s = 0; s < 2; ++s) {
        const int c = tid + s * 256;               // chunk 0..511 (16B each)
        const unsigned loff =
            (unsigned)(uintptr_t)lds_tile + (unsigned)((c >> 5) * (BROW_H * 2) + (c & 31) * 16);
        const void* g = (const char*)gsrc + (size_t)c * 16;
        asm volatile("global_load_async_to_lds_b128 %0, %1, off"
                     :: "v"(loff), "v"(g) : "memory");
    }
}

// ---------------------------------------------------------------------------
// Kernel 2: fused sim = reps @ reps^T via v_wmma_f32_16x16x32_f16.
// Block = 8 waves; each wave owns a 16-row i-tile; all waves share the same
// j sweep, so the B tile is staged once per block in double-buffered LDS by
// the async tensor path, then read with ds_load_b128 in this code's contiguous K order
// (lane L holds col N=L%16, K=16*(L/16)+0..15 -> 32 contiguous bytes of the
// reps row). A fragments (K=256 -> 8 x v16h) live in registers for the whole
// sweep. exp / mask / row-sum fused per 16x16 tile; partner (numerator)
// element captured only in the single uniform partner tile (cj0 == mi0^4096).
// ---------------------------------------------------------------------------
__global__ void cl_wmma_loss_kernel(const _Float16* __restrict__ reps,
                                    const unsigned* __restrict__ meta,
                                    float* __restrict__ denom,
                                    float* __restrict__ numer) {
    __shared__ __align__(16) _Float16 btile[2][BTILE_H];

    const int tid    = threadIdx.x;
    const int lane   = tid & 31;
    const int waveId = tid >> 5;
    const int nIdx   = lane & 15;
    const int laneHi = lane >> 4;                 // 0 or 1

    const int itile = blockIdx.x * 8 + waveId;    // 0..511
    const int mi0   = itile * 16;
    const int pj0   = mi0 ^ BN;                   // partner tile column base
    const int j0    = blockIdx.y * (TWO_B / JSPLIT);

    // ---- A fragments, loaded once, contiguous K=16*(L/16)+0..15 (matches B) ----
    const _Float16* arow = reps + (size_t)(mi0 + nIdx) * DDIM;
    v16h afrag[8];
#pragma unroll
    for (int t = 0; t < 8; ++t) {
        const int k0 = t * 32;
        afrag[t] = *(const v16h*)(arow + k0 + 16 * laneHi);
    }

    // ---- Packed (tag<<16)|doc for the 8 output rows this lane owns ----
    unsigned mi[8];
#pragma unroll
    for (int r = 0; r < 8; ++r) mi[r] = meta[mi0 + r + 8 * laneHi];

    float accum[8];
#pragma unroll
    for (int r = 0; r < 8; ++r) accum[r] = 0.0f;

    // ---- Prologue: stage tile 0 ----
    issue_btile(reps + (size_t)j0 * DDIM, &btile[0][0], tid);
    asm volatile("s_wait_asynccnt 0x0" ::: "memory");
    __syncthreads();

    for (int jt = 0; jt < JTILES; ++jt) {
        const int p = jt & 1;

        // Prefetch next tile into the other buffer (overlaps WMMA chain).
        if (jt + 1 < JTILES)
            issue_btile(reps + (size_t)(j0 + (jt + 1) * 16) * DDIM,
                        &btile[p ^ 1][0], tid);

        const int cj0 = j0 + jt * 16;
        const int gj  = cj0 + nIdx;               // this lane's output column
        const unsigned mj = meta[gj];

        // ---- B fragments from LDS (ds_load_b128 pairs), then WMMA chain ----
        const _Float16* bt = &btile[p][nIdx * BROW_H + 16 * laneHi];
        v16h bfrag[8];
#pragma unroll
        for (int t = 0; t < 8; ++t)
            bfrag[t] = *(const v16h*)(bt + t * 32);

        v8f d = {};
#pragma unroll
        for (int t = 0; t < 8; ++t)
            d = __builtin_amdgcn_wmma_f32_16x16x32_f16(
                    false, afrag[t], false, bfrag[t],
                    (short)0, d, false, false);

        // ---- exp(sim/0.5) = exp(2*sim); mask; accumulate ----
#pragma unroll
        for (int r = 0; r < 8; ++r) {
            const float    e = __expf(d[r] * 2.0f);     // v_exp_f32 (trans)
            const unsigned x = mi[r] ^ mj;
            const bool diff  = ((x >> 16) != 0u) & ((x & 0xFFFFu) != 0u);
            accum[r] += diff ? e : 0.0f;
        }

        // ---- Numerator: only in the (uniform) partner tile, diagonal ----
        if (cj0 == pj0) {
            if ((nIdx >> 3) == laneHi) {
                const int r = nIdx & 7;
                float sd = d[0];
#pragma unroll
                for (int q = 1; q < 8; ++q) sd = (r == q) ? d[q] : sd;
                numer[mi0 + nIdx] = __expf(sd * 2.0f);
            }
        }

        // Our async loads into buffer p^1 must land before anyone reads it,
        // and everyone must be done reading buffer p before it is reused.
        asm volatile("s_wait_asynccnt 0x0" ::: "memory");
        __syncthreads();
    }

    // ---- Reduce accum over the 16 lanes sharing each row, then atomic ----
#pragma unroll
    for (int r = 0; r < 8; ++r) {
        float s = accum[r];
        s += __shfl_xor(s, 1, 32);
        s += __shfl_xor(s, 2, 32);
        s += __shfl_xor(s, 4, 32);
        s += __shfl_xor(s, 8, 32);        // halves (0-15 / 16-31) stay separate
        if (nIdx == 0)
            atomicAdd(&denom[mi0 + r + 8 * laneHi], s);   // global_atomic_add_f32
    }
}

// ---------------------------------------------------------------------------
// Kernel 3: loss = mean_i [ log(denom_i + 0.1) - log(numer_i) ]
// ---------------------------------------------------------------------------
__global__ void cl_finalize_kernel(const float* __restrict__ denom,
                                   const float* __restrict__ numer,
                                   float* __restrict__ out) {
    __shared__ float sdata[256];
    float s = 0.0f;
    for (int i = threadIdx.x; i < TWO_B; i += 256)
        s += __logf(denom[i] + 0.1f) - __logf(numer[i]);  // v_log_f32
    sdata[threadIdx.x] = s;
    __syncthreads();
    for (int off = 128; off > 0; off >>= 1) {
        if ((int)threadIdx.x < off) sdata[threadIdx.x] += sdata[threadIdx.x + off];
        __syncthreads();
    }
    if (threadIdx.x == 0) out[0] = sdata[0] / (float)TWO_B;
}

// ---------------------------------------------------------------------------
// Launch
// ---------------------------------------------------------------------------
extern "C" void kernel_launch(void* const* d_in, const int* in_sizes, int n_in,
                              void* d_out, int out_size, void* d_ws, size_t ws_size,
                              hipStream_t stream) {
    const float* emb_i = (const float*)d_in[0];   // [4096,256] f32
    const float* emb_j = (const float*)d_in[1];   // [4096,256] f32
    const int*   tags  = (const int*)d_in[2];     // [4096]
    // d_in[3] = num_classes (unused)
    const int*   docs  = (const int*)d_in[4];     // [4096]
    float*       out   = (float*)d_out;

    // Workspace: f16 reps (4MB) | meta (32KB) | denom (32KB) | numer (32KB)
    char*     ws    = (char*)d_ws;
    _Float16* reps  = (_Float16*)ws;
    unsigned* meta  = (unsigned*)(ws + (size_t)TWO_B * DDIM * sizeof(_Float16));
    float*    denom = (float*)(meta + TWO_B);
    float*    numer = denom + TWO_B;

    // 1) normalize + metadata pack (one wave per row)
    cl_normalize_kernel<<<dim3((TWO_B * 32) / 256), dim3(256), 0, stream>>>(
        emb_i, emb_j, tags, docs, reps, meta, denom, numer);

    // 2) fused WMMA GEMM + loss terms: 64 blocks x JSPLIT, 8 waves each.
    cl_wmma_loss_kernel<<<dim3(512 / 8, JSPLIT), dim3(256), 0, stream>>>(
        reps, meta, denom, numer);

    // 3) final reduction to scalar loss
    cl_finalize_kernel<<<dim3(1), dim3(256), 0, stream>>>(denom, numer, out);
}
